// GNNEncoderLayer_69578470195890
// MI455X (gfx1250) — compile-verified, matrix operand order checked by reading, not run on hardware
//
#include <hip/hip_runtime.h>

#define NN 50000
#define NE 1500000
#define DIM 512

typedef __attribute__((ext_vector_type(16))) __bf16 v16bf;
typedef __attribute__((ext_vector_type(8)))  float  v8f;

union Pack16 { uint4 u[2]; v16bf v; __bf16 b[16]; };
union F16x   { float4 f[4]; float a[16]; };

// ---------------- zero workspace ----------------
__global__ void k_zero(float* __restrict__ p, long long n) {
    long long i = ((long long)blockIdx.x * blockDim.x + threadIdx.x) * 4;
    if (i + 3 < n) {
        *(float4*)(p + i) = make_float4(0.f, 0.f, 0.f, 0.f);
    } else {
        for (; i < n; ++i) p[i] = 0.f;
    }
}

// ---------------- LayerNorm -> bf16 (one wave per row) ----------------
__global__ void k_ln(const float* __restrict__ x, const float* __restrict__ gamma,
                     const float* __restrict__ beta, __bf16* __restrict__ hbf) {
    int wid  = (int)(((unsigned)blockIdx.x * blockDim.x + threadIdx.x) >> 5);
    int lane = threadIdx.x & 31;
    if (wid >= NN) return;

    const float* rp = x + (size_t)wid * DIM + lane * 16;
    F16x v;
#pragma unroll
    for (int j = 0; j < 4; ++j) v.f[j] = *(const float4*)(rp + 4 * j);

    float s = 0.f, ss = 0.f;
#pragma unroll
    for (int i = 0; i < 16; ++i) { s += v.a[i]; ss += v.a[i] * v.a[i]; }
#pragma unroll
    for (int off = 16; off > 0; off >>= 1) {
        s  += __shfl_xor(s,  off, 32);
        ss += __shfl_xor(ss, off, 32);
    }
    float mu = s * (1.0f / DIM);
    float var = ss * (1.0f / DIM) - mu * mu;
    float rs = rsqrtf(var + 1e-5f);

    F16x g, b;
    const float* gp = gamma + lane * 16;
    const float* bp = beta  + lane * 16;
#pragma unroll
    for (int j = 0; j < 4; ++j) { g.f[j] = *(const float4*)(gp + 4 * j);
                                  b.f[j] = *(const float4*)(bp + 4 * j); }
    Pack16 o;
#pragma unroll
    for (int i = 0; i < 16; ++i)
        o.b[i] = (__bf16)((v.a[i] - mu) * rs * g.a[i] + b.a[i]);

    uint4* op = (uint4*)(hbf + (size_t)wid * DIM + lane * 16);
    op[0] = o.u[0];
    op[1] = o.u[1];
}

// ---------------- weights f32 -> bf16 ----------------
__global__ void k_wcvt(const float* __restrict__ wl, const float* __restrict__ wr,
                       __bf16* __restrict__ wlb, __bf16* __restrict__ wrb) {
    int i = blockIdx.x * blockDim.x + threadIdx.x;
    if (i < DIM * DIM) {
        wlb[i] = (__bf16)wl[i];
        wrb[i] = (__bf16)wr[i];
    }
}

// ---------------- edge scatter: one wave per edge ----------------
__global__ void k_edge(const int* __restrict__ ei, const __bf16* __restrict__ hbf,
                       float* __restrict__ summed, float* __restrict__ cnt) {
    long long t = (long long)blockIdx.x * blockDim.x + threadIdx.x;
    int e    = (int)(t >> 5);
    int lane = threadIdx.x & 31;
    if (e >= NE) return;

    int s = ei[e];        // src
    int d = ei[NE + e];   // dst

    const uint4* hp = (const uint4*)(hbf + (size_t)s * DIM + lane * 16);
    Pack16 p;
    p.u[0] = hp[0];
    p.u[1] = hp[1];

    float* op = summed + (size_t)d * DIM + lane * 16;
#pragma unroll
    for (int i = 0; i < 16; ++i)
        unsafeAtomicAdd(op + i, (float)p.b[i]);
    if (lane == 0) unsafeAtomicAdd(cnt + d, 1.0f);
}

// ---------------- mean = summed / max(cnt,1) -> bf16 ----------------
__global__ void k_mean(const float* __restrict__ summed, const float* __restrict__ cnt,
                       __bf16* __restrict__ meanbf) {
    int t = blockIdx.x * blockDim.x + threadIdx.x;   // one thread per 16 elems
    int row = t >> 5;                                // 32 threads per row
    int seg = (t & 31) * 16;
    if (row >= NN) return;
    float inv = 1.0f / fmaxf(cnt[row], 1.0f);
    const float* sp = summed + (size_t)row * DIM + seg;
    F16x v;
#pragma unroll
    for (int j = 0; j < 4; ++j) v.f[j] = *(const float4*)(sp + 4 * j);
    Pack16 o;
#pragma unroll
    for (int i = 0; i < 16; ++i) o.b[i] = (__bf16)(v.a[i] * inv);
    uint4* op = (uint4*)(meanbf + (size_t)row * DIM + seg);
    op[0] = o.u[0];
    op[1] = o.u[1];
}

// ---------------- fused dual-GEMM + bias + ReLU + residual ----------------
// One wave computes a 16x16 output tile. acc += mean@Wl^T + h@Wr^T via
// v_wmma_f32_16x16x32_bf16, K swept in steps of 32.
__global__ __launch_bounds__(256) void k_gemm(
        const __bf16* __restrict__ meanbf, const __bf16* __restrict__ hbf,
        const __bf16* __restrict__ wlb,    const __bf16* __restrict__ wrb,
        const float*  __restrict__ bias,   const float* __restrict__ x,
        float* __restrict__ out) {
    int wid  = (int)(((unsigned)blockIdx.x * blockDim.x + threadIdx.x) >> 5);
    int lane = threadIdx.x & 31;
    int mt = wid >> 5;          // row tile  0..3124
    int nt = wid & 31;          // col tile  0..31
    if (mt >= NN / 16) return;

    int r   = lane & 15;
    int hid = lane >> 4;        // which K-half this lane holds
    int arow = mt * 16 + r;     // A: lane = output row
    int bcol = nt * 16 + r;     // B: lane = output col (W row-major [n][k])

    const __bf16* ap = meanbf + (size_t)arow * DIM;
    const __bf16* hp = hbf    + (size_t)arow * DIM;
    const __bf16* blp = wlb + (size_t)bcol * DIM;
    const __bf16* brp = wrb + (size_t)bcol * DIM;

    v8f acc = {};
    for (int k0 = 0; k0 < DIM; k0 += 32) {
        int ka = k0 + hid * 8;  // 16-bit layout: lanes 0-15 K=[0..7],[16..23]; 16-31 +8
        Pack16 a1, a2, b1, b2;
        a1.u[0] = *(const uint4*)(ap + ka);      a1.u[1] = *(const uint4*)(ap + ka + 16);
        a2.u[0] = *(const uint4*)(hp + ka);      a2.u[1] = *(const uint4*)(hp + ka + 16);
        b1.u[0] = *(const uint4*)(blp + ka);     b1.u[1] = *(const uint4*)(blp + ka + 16);
        b2.u[0] = *(const uint4*)(brp + ka);     b2.u[1] = *(const uint4*)(brp + ka + 16);

        acc = __builtin_amdgcn_wmma_f32_16x16x32_bf16(false, a1.v, false, b1.v,
                                                      (short)0, acc, false, false);
        acc = __builtin_amdgcn_wmma_f32_16x16x32_bf16(false, a2.v, false, b2.v,
                                                      (short)0, acc, false, false);
    }

    // D layout: VGPR i -> row mt*16 + hid*8 + i, col nt*16 + (lane&15)
    int col   = nt * 16 + r;
    int mbase = mt * 16 + hid * 8;
    float bv = bias[col];
#pragma unroll
    for (int i = 0; i < 8; ++i) {
        int row = mbase + i;
        float v = acc[i] + bv;
        v = fmaxf(v, 0.0f) + x[(size_t)row * DIM + col];
        out[(size_t)row * DIM + col] = v;
    }
}

extern "C" void kernel_launch(void* const* d_in, const int* in_sizes, int n_in,
                              void* d_out, int out_size, void* d_ws, size_t ws_size,
                              hipStream_t stream) {
    const float* x     = (const float*)d_in[0];
    const int*   ei    = (const int*)  d_in[1];
    const float* Wl    = (const float*)d_in[2];
    const float* bl    = (const float*)d_in[3];
    const float* Wr    = (const float*)d_in[4];
    const float* gamma = (const float*)d_in[5];
    const float* beta  = (const float*)d_in[6];
    float* out = (float*)d_out;

    char* ws = (char*)d_ws;
    size_t off_h    = 0;                                   // h bf16:    N*D*2
    size_t off_sum  = off_h    + (size_t)NN * DIM * 2;     // summed f32: N*D*4
    size_t off_cnt  = off_sum  + (size_t)NN * DIM * 4;     // cnt f32:    N*4
    size_t off_mean = off_cnt  + (size_t)NN * 4;           // mean bf16:  N*D*2
    size_t off_wl   = off_mean + (size_t)NN * DIM * 2;     // Wl bf16
    size_t off_wr   = off_wl   + (size_t)DIM * DIM * 2;    // Wr bf16

    __bf16* hbf    = (__bf16*)(ws + off_h);
    float*  summed = (float*) (ws + off_sum);
    float*  cnt    = (float*) (ws + off_cnt);
    __bf16* meanbf = (__bf16*)(ws + off_mean);
    __bf16* wlb    = (__bf16*)(ws + off_wl);
    __bf16* wrb    = (__bf16*)(ws + off_wr);

    // 1. zero accumulators (summed + cnt are contiguous)
    long long nz = (long long)NN * DIM + NN;
    k_zero<<<(int)((nz / 4 + 255) / 256), 256, 0, stream>>>(summed, nz);

    // 2. LayerNorm -> bf16 (one wave per row: 50000 waves)
    k_ln<<<NN * 32 / 256, 256, 0, stream>>>(x, gamma, beta, hbf);

    // 3. weights -> bf16
    k_wcvt<<<(DIM * DIM) / 256, 256, 0, stream>>>(Wl, Wr, wlb, wrb);

    // 4. edge gather + atomic scatter (one wave per edge)
    k_edge<<<NE * 32 / 256, 256, 0, stream>>>(ei, hbf, summed, cnt);

    // 5. mean -> bf16
    k_mean<<<NN * 32 / 256, 256, 0, stream>>>(summed, cnt, meanbf);

    // 6. fused dual WMMA GEMM + epilogue (3125*32 tiles, 8 waves/block)
    k_gemm<<<(NN / 16) * 32 / 8, 256, 0, stream>>>(meanbf, hbf, wlb, wrb, bl, x, out);
}
